// LinearGLUMoELayer_15307263443374
// MI455X (gfx1250) — matrix operand order repaired and checked by reading, not compiled, not run on hardware
//
#include <hip/hip_runtime.h>

typedef __attribute__((ext_vector_type(16))) __bf16 bf16x16;
typedef __attribute__((ext_vector_type(8)))  float  f32x8;

#define E_NUM 8
#define TM 128
#define TN 128
#define TK 32
#define LDSTR 40   // ushort stride -> 80 bytes, keeps 16B alignment for b128 LDS ops

union FragAB { bf16x16 v; uint4 q[2]; };

__device__ __forceinline__ unsigned short f2bf(float f) {
    unsigned u = __float_as_uint(f);
    u = u + 0x7FFFu + ((u >> 16) & 1u);   // RNE
    return (unsigned short)(u >> 16);
}

__device__ __forceinline__ f32x8 zero8() {
    f32x8 z;
#pragma unroll
    for (int i = 0; i < 8; ++i) z[i] = 0.f;
    return z;
}

// CDNA5 async global->LDS DMA (ASYNCcnt-tracked, bypasses VGPRs).
// lds_off: per-lane LDS byte address (low 32 bits of generic pointer),
// gptr:    per-lane 64-bit global address.
__device__ __forceinline__ void async_ld128(unsigned lds_off, const void* gptr) {
    asm volatile("global_load_async_to_lds_b128 %0, %1, off"
                 :: "v"(lds_off), "v"(gptr) : "memory");
}
__device__ __forceinline__ void wait_async0() {
    asm volatile("s_wait_asynccnt 0" ::: "memory");
}

__device__ __forceinline__ unsigned lds_off_u32(const void* p) {
    return (unsigned)(unsigned long long)p;
}

// ---------------- small utility kernels ----------------

__global__ void k_zero_i32(int* p, int n) {
    int i = blockIdx.x * blockDim.x + threadIdx.x;
    if (i < n) p[i] = 0;
}

// fp32 -> bf16, 4 elements per thread
__global__ void k_cvt_bf16(const float* __restrict__ src,
                           unsigned short* __restrict__ dst, long long n4) {
    long long i = (long long)blockIdx.x * blockDim.x + threadIdx.x;
    if (i >= n4) return;
    float4 f = ((const float4*)src)[i];
    ushort4 o;
    o.x = f2bf(f.x); o.y = f2bf(f.y); o.z = f2bf(f.z); o.w = f2bf(f.w);
    ((ushort4*)dst)[i] = o;
}

// src: [E][R][C] fp32  ->  dst: [E][C][R] bf16  (K-contiguous weight layout)
__global__ void k_transpose_bf16(const float* __restrict__ src,
                                 unsigned short* __restrict__ dst, int R, int C) {
    __shared__ float tile[32][33];
    int e  = blockIdx.z;
    int c0 = blockIdx.x * 32;
    int r0 = blockIdx.y * 32;
    const float* s = src + (size_t)e * R * C;
    unsigned short* d = dst + (size_t)e * R * C;
    for (int i = threadIdx.y; i < 32; i += 8) {
        int r = r0 + i, c = c0 + threadIdx.x;
        tile[i][threadIdx.x] = (r < R && c < C) ? s[(size_t)r * C + c] : 0.f;
    }
    __syncthreads();
    for (int i = threadIdx.y; i < 32; i += 8) {
        int c = c0 + i, r = r0 + threadIdx.x;
        if (c < C && r < R) d[(size_t)c * R + r] = f2bf(tile[threadIdx.x][i]);
    }
}

// ---------------- gating (fp32, matches reference numerics) ----------------

__global__ void k_gate(const float* __restrict__ x, const float* __restrict__ gw1,
                       const float* __restrict__ gw2,
                       float* __restrict__ scoreA, int* __restrict__ idxA,
                       int* __restrict__ cnt, int N, int D) {
    int n = blockIdx.x * blockDim.x + threadIdx.x;
    if (n >= N) return;
    const float* xr = x + (size_t)n * D;
    float acc[E_NUM];
#pragma unroll
    for (int j = 0; j < E_NUM; ++j) acc[j] = 0.f;
    for (int d = 0; d < D; d += 4) {
        float4 xv = *(const float4*)(xr + d);
#pragma unroll
        for (int j = 0; j < E_NUM; ++j) {
            float4 w = *(const float4*)(gw1 + (size_t)j * D + d);
            acc[j] += xv.x * w.x + xv.y * w.y + xv.z * w.z + xv.w * w.w;
        }
    }
    float t[E_NUM], lg[E_NUM];
#pragma unroll
    for (int j = 0; j < E_NUM; ++j) t[j] = tanhf(acc[j]);
#pragma unroll
    for (int i = 0; i < E_NUM; ++i) {
        float s = 0.f;
#pragma unroll
        for (int j = 0; j < E_NUM; ++j) s += gw2[i * E_NUM + j] * t[j];
        lg[i] = s;
    }
    // top-2, tie-break lower index first (strict >)
    int i0 = 0; float v0 = lg[0];
#pragma unroll
    for (int i = 1; i < E_NUM; ++i) if (lg[i] > v0) { v0 = lg[i]; i0 = i; }
    int i1 = -1; float v1 = -3.0e38f;
#pragma unroll
    for (int i = 0; i < E_NUM; ++i) if (i != i0 && lg[i] > v1) { v1 = lg[i]; i1 = i; }
    float e1 = __expf(v1 - v0);
    float inv = 1.f / (1.f + e1);
    scoreA[2 * n]     = inv;
    scoreA[2 * n + 1] = e1 * inv;
    idxA[2 * n]     = i0;
    idxA[2 * n + 1] = i1;
    atomicAdd(&cnt[i0], 1);
    atomicAdd(&cnt[i1], 1);
}

__global__ void k_offsets(const int* __restrict__ cnt, int* __restrict__ offs,
                          int* __restrict__ cursor) {
    if (threadIdx.x == 0) {
        int s = 0;
        for (int e = 0; e < E_NUM; ++e) { offs[e] = s; cursor[e] = s; s += cnt[e]; }
    }
}

__global__ void k_fill(const float* __restrict__ scoreA, const int* __restrict__ idxA,
                       int* __restrict__ cursor, int* __restrict__ ptok,
                       int* __restrict__ pslot, float* __restrict__ pscore, int N) {
    int n = blockIdx.x * blockDim.x + threadIdx.x;
    if (n >= N) return;
#pragma unroll
    for (int s = 0; s < 2; ++s) {
        int e = idxA[2 * n + s];
        int p = atomicAdd(&cursor[e], 1);
        ptok[p] = n; pslot[p] = s; pscore[p] = scoreA[2 * n + s];
    }
}

// ---------------- grouped GLU GEMM: h = silu(X*Wg) * (X*Wu) ----------------
// A: gathered token rows (bf16 [N][D]); B: wgT/wuT [E][H][D] (K=D contiguous)
// Double-buffered LDS, tiles fetched with global_load_async_to_lds_b128.

__global__ __launch_bounds__(256)
void k_glu_gemm(const unsigned short* __restrict__ xb,
                const unsigned short* __restrict__ wgT,
                const unsigned short* __restrict__ wuT,
                const int* __restrict__ ptok,
                const int* __restrict__ cnt, const int* __restrict__ offs,
                unsigned short* __restrict__ hbuf, int D, int H) {
    int e = blockIdx.z;
    int m = cnt[e];
    int rt = blockIdx.y;
    if (rt * TM >= m) return;
    int ct = blockIdx.x;
    int base = offs[e];

    __shared__ __align__(16) unsigned short sA[2][TM * LDSTR];
    __shared__ __align__(16) unsigned short sG[2][TN * LDSTR];
    __shared__ __align__(16) unsigned short sU[2][TN * LDSTR];

    const int tid  = threadIdx.x;
    const int lane = tid & 31;
    const int wave = tid >> 5;
    const int wm   = wave & 3;   // 4 row-groups of 32 rows
    const int wn   = wave >> 2;  // 2 col-groups of 64 cols
    const int r16  = lane & 15;
    const int hsel = lane >> 4;

    // per-thread load assignment: chunks of 8 bf16 (16B); 512 chunks -> 2/thread
    const int row0 = tid >> 2;            // 0..63
    const int row1 = (tid + 256) >> 2;    // 64..127
    const int kc   = (tid & 3) * 8;       // same for both chunks (256 % 4 == 0)

    int rg0 = rt * TM + row0; if (rg0 >= m) rg0 = m - 1;
    int rg1 = rt * TM + row1; if (rg1 >= m) rg1 = m - 1;
    const unsigned short* pa0 = xb + (size_t)ptok[base + rg0] * D + kc;
    const unsigned short* pa1 = xb + (size_t)ptok[base + rg1] * D + kc;
    const unsigned short* pg0 = wgT + ((size_t)e * H + ct * TN + row0) * D + kc;
    const unsigned short* pg1 = wgT + ((size_t)e * H + ct * TN + row1) * D + kc;
    const unsigned short* pu0 = wuT + ((size_t)e * H + ct * TN + row0) * D + kc;
    const unsigned short* pu1 = wuT + ((size_t)e * H + ct * TN + row1) * D + kc;

    const unsigned oA0 = lds_off_u32(&sA[0][row0 * LDSTR + kc]);
    const unsigned oA1 = lds_off_u32(&sA[0][row1 * LDSTR + kc]);
    const unsigned oG0 = lds_off_u32(&sG[0][row0 * LDSTR + kc]);
    const unsigned oG1 = lds_off_u32(&sG[0][row1 * LDSTR + kc]);
    const unsigned oU0 = lds_off_u32(&sU[0][row0 * LDSTR + kc]);
    const unsigned oU1 = lds_off_u32(&sU[0][row1 * LDSTR + kc]);
    const unsigned BUF = (unsigned)(TM * LDSTR * 2);   // bytes between buffers

    // prologue: async DMA of tile 0 into buffer 0
    async_ld128(oA0, pa0); async_ld128(oA1, pa1);
    async_ld128(oG0, pg0); async_ld128(oG1, pg1);
    async_ld128(oU0, pu0); async_ld128(oU1, pu1);
    pa0 += TK; pa1 += TK; pg0 += TK; pg1 += TK; pu0 += TK; pu1 += TK;

    f32x8 cg[2][4], cu[2][4];
#pragma unroll
    for (int mi = 0; mi < 2; ++mi)
#pragma unroll
        for (int ni = 0; ni < 4; ++ni) { cg[mi][ni] = zero8(); cu[mi][ni] = zero8(); }

    const int KIT = D / TK;
    for (int kt = 0; kt < KIT; ++kt) {
        wait_async0();       // my wave's DMA for the current buffer has landed
        __syncthreads();     // everyone drained -> buffer visible; prev reads done
        const unsigned cb = (unsigned)(kt & 1);
        if (kt + 1 < KIT) {  // DMA next tile into the other buffer
            const unsigned nb = (cb ^ 1u) * BUF;
            async_ld128(oA0 + nb, pa0); async_ld128(oA1 + nb, pa1);
            async_ld128(oG0 + nb, pg0); async_ld128(oG1 + nb, pg1);
            async_ld128(oU0 + nb, pu0); async_ld128(oU1 + nb, pu1);
            pa0 += TK; pa1 += TK; pg0 += TK; pg1 += TK; pu0 += TK; pu1 += TK;
        }
        const unsigned short* sAb = sA[cb];
        const unsigned short* sGb = sG[cb];
        const unsigned short* sUb = sU[cb];

        FragAB a[2];
#pragma unroll
        for (int mi = 0; mi < 2; ++mi) {
            const unsigned short* p = sAb + (wm * 32 + mi * 16 + r16) * LDSTR + hsel * 8;
            a[mi].q[0] = *(const uint4*)(p);
            a[mi].q[1] = *(const uint4*)(p + 16);
        }
#pragma unroll
        for (int ni = 0; ni < 4; ++ni) {
            const int nrow = (wn * 64 + ni * 16 + r16) * LDSTR + hsel * 8;
            FragAB bg, bu;
            bg.q[0] = *(const uint4*)(sGb + nrow);
            bg.q[1] = *(const uint4*)(sGb + nrow + 16);
            bu.q[0] = *(const uint4*)(sUb + nrow);
            bu.q[1] = *(const uint4*)(sUb + nrow + 16);
#pragma unroll
            for (int mi = 0; mi < 2; ++mi) {
                cg[mi][ni] = __builtin_amdgcn_wmma_f32_16x16x32_bf16(
                    false, a[mi].v, false, bg.v, (short)0, cg[mi][ni], false, false);
                cu[mi][ni] = __builtin_amdgcn_wmma_f32_16x16x32_bf16(
                    false, a[mi].v, false, bu.v, (short)0, cu[mi][ni], false, false);
            }
        }
    }

    // epilogue: h = silu(g) * u  -> bf16
#pragma unroll
    for (int mi = 0; mi < 2; ++mi) {
#pragma unroll
        for (int i = 0; i < 8; ++i) {
            int rloc = wm * 32 + mi * 16 + i + hsel * 8;
            int r = rt * TM + rloc;
            if (r < m) {
                size_t rowoff = (size_t)(base + r) * H + ct * TN + wn * 64 + r16;
#pragma unroll
                for (int ni = 0; ni < 4; ++ni) {
                    float g = cg[mi][ni][i], u = cu[mi][ni][i];
                    float hv = (g / (1.f + __expf(-g))) * u;
                    hbuf[rowoff + ni * 16] = f2bf(hv);
                }
            }
        }
    }
}

// ---------------- grouped down-proj GEMM: out2[token][slot] = score * (h * Wd) ----

__global__ __launch_bounds__(256)
void k_down_gemm(const unsigned short* __restrict__ hbuf,   // [2N][H]
                 const unsigned short* __restrict__ wdT,    // [E][D][H]
                 const int* __restrict__ ptok, const int* __restrict__ pslot,
                 const float* __restrict__ pscore,
                 const int* __restrict__ cnt, const int* __restrict__ offs,
                 float* __restrict__ out2, int D, int H) {
    int e = blockIdx.z;
    int m = cnt[e];
    int rt = blockIdx.y;
    if (rt * TM >= m) return;
    int ct = blockIdx.x;
    int base = offs[e];

    __shared__ __align__(16) unsigned short sA[2][TM * LDSTR];
    __shared__ __align__(16) unsigned short sB[2][TN * LDSTR];

    const int tid  = threadIdx.x;
    const int lane = tid & 31;
    const int wave = tid >> 5;
    const int wm   = wave & 3;
    const int wn   = wave >> 2;
    const int r16  = lane & 15;
    const int hsel = lane >> 4;

    const int row0 = tid >> 2;
    const int row1 = (tid + 256) >> 2;
    const int kc   = (tid & 3) * 8;

    int rg0 = rt * TM + row0; if (rg0 >= m) rg0 = m - 1;
    int rg1 = rt * TM + row1; if (rg1 >= m) rg1 = m - 1;
    const unsigned short* pa0 = hbuf + (size_t)(base + rg0) * H + kc;
    const unsigned short* pa1 = hbuf + (size_t)(base + rg1) * H + kc;
    const unsigned short* pb0 = wdT + ((size_t)e * D + ct * TN + row0) * H + kc;
    const unsigned short* pb1 = wdT + ((size_t)e * D + ct * TN + row1) * H + kc;

    const unsigned oA0 = lds_off_u32(&sA[0][row0 * LDSTR + kc]);
    const unsigned oA1 = lds_off_u32(&sA[0][row1 * LDSTR + kc]);
    const unsigned oB0 = lds_off_u32(&sB[0][row0 * LDSTR + kc]);
    const unsigned oB1 = lds_off_u32(&sB[0][row1 * LDSTR + kc]);
    const unsigned BUF = (unsigned)(TM * LDSTR * 2);

    async_ld128(oA0, pa0); async_ld128(oA1, pa1);
    async_ld128(oB0, pb0); async_ld128(oB1, pb1);
    pa0 += TK; pa1 += TK; pb0 += TK; pb1 += TK;

    f32x8 c[2][4];
#pragma unroll
    for (int mi = 0; mi < 2; ++mi)
#pragma unroll
        for (int ni = 0; ni < 4; ++ni) c[mi][ni] = zero8();

    const int KIT = H / TK;
    for (int kt = 0; kt < KIT; ++kt) {
        wait_async0();
        __syncthreads();
        const unsigned cb = (unsigned)(kt & 1);
        if (kt + 1 < KIT) {
            const unsigned nb = (cb ^ 1u) * BUF;
            async_ld128(oA0 + nb, pa0); async_ld128(oA1 + nb, pa1);
            async_ld128(oB0 + nb, pb0); async_ld128(oB1 + nb, pb1);
            pa0 += TK; pa1 += TK; pb0 += TK; pb1 += TK;
        }
        const unsigned short* sAb = sA[cb];
        const unsigned short* sBb = sB[cb];

        FragAB a[2];
#pragma unroll
        for (int mi = 0; mi < 2; ++mi) {
            const unsigned short* p = sAb + (wm * 32 + mi * 16 + r16) * LDSTR + hsel * 8;
            a[mi].q[0] = *(const uint4*)(p);
            a[mi].q[1] = *(const uint4*)(p + 16);
        }
#pragma unroll
        for (int ni = 0; ni < 4; ++ni) {
            const int nrow = (wn * 64 + ni * 16 + r16) * LDSTR + hsel * 8;
            FragAB b;
            b.q[0] = *(const uint4*)(sBb + nrow);
            b.q[1] = *(const uint4*)(sBb + nrow + 16);
#pragma unroll
            for (int mi = 0; mi < 2; ++mi)
                c[mi][ni] = __builtin_amdgcn_wmma_f32_16x16x32_bf16(
                    false, a[mi].v, false, b.v, (short)0, c[mi][ni], false, false);
        }
    }

#pragma unroll
    for (int mi = 0; mi < 2; ++mi) {
#pragma unroll
        for (int i = 0; i < 8; ++i) {
            int rloc = wm * 32 + mi * 16 + i + hsel * 8;
            int r = rt * TM + rloc;
            if (r < m) {
                int p = base + r;
                int tok = ptok[p], sl = pslot[p];
                float sc = pscore[p];
                size_t o = ((size_t)tok * 2 + sl) * D + ct * TN + wn * 64 + r16;
#pragma unroll
                for (int ni = 0; ni < 4; ++ni) out2[o + ni * 16] = c[mi][ni][i] * sc;
            }
        }
    }
}

// ---------------- combine + loss ----------------

__global__ void k_combine(const float* __restrict__ out2, float* __restrict__ out,
                          int N, int D) {
    long long i = (long long)blockIdx.x * blockDim.x + threadIdx.x;
    int d4 = D / 4;
    long long total = (long long)N * d4;
    if (i >= total) return;
    int n = (int)(i / d4), dd = (int)(i % d4);
    float4 va = ((const float4*)out2)[(size_t)(2 * n) * d4 + dd];
    float4 vb = ((const float4*)out2)[(size_t)(2 * n + 1) * d4 + dd];
    float4 r; r.x = va.x + vb.x; r.y = va.y + vb.y; r.z = va.z + vb.z; r.w = va.w + vb.w;
    ((float4*)out)[(size_t)n * d4 + dd] = r;
}

__global__ void k_loss(const float* __restrict__ scoreA, const int* __restrict__ idxA,
                       const int* __restrict__ cnt, float* __restrict__ out_loss, int N) {
    __shared__ float simp[256][E_NUM];
    int t = threadIdx.x;
#pragma unroll
    for (int e = 0; e < E_NUM; ++e) simp[t][e] = 0.f;
    for (int n = t; n < N; n += 256) {
        simp[t][idxA[2 * n]]     += scoreA[2 * n];
        simp[t][idxA[2 * n + 1]] += scoreA[2 * n + 1];
    }
    __syncthreads();
    for (int s = 128; s > 0; s >>= 1) {
        if (t < s)
#pragma unroll
            for (int e = 0; e < E_NUM; ++e) simp[t][e] += simp[t + s][e];
        __syncthreads();
    }
    if (t == 0) {
        float mi = 0.f;
        for (int e = 0; e < E_NUM; ++e) mi += simp[0][e];
        mi /= E_NUM;
        float vi = 0.f;
        for (int e = 0; e < E_NUM; ++e) { float d = simp[0][e] - mi; vi += d * d; }
        vi /= (E_NUM - 1);
        float ml = 0.f;
        for (int e = 0; e < E_NUM; ++e) ml += (float)cnt[e];
        ml /= E_NUM;
        float vl = 0.f;
        for (int e = 0; e < E_NUM; ++e) { float d = (float)cnt[e] - ml; vl += d * d; }
        vl /= (E_NUM - 1);
        *out_loss = 0.01f * (vi / (mi * mi + 1e-10f) + vl / (ml * ml + 1e-10f));
    }
}

// ---------------- host launcher ----------------

extern "C" void kernel_launch(void* const* d_in, const int* in_sizes, int n_in,
                              void* d_out, int out_size, void* d_ws, size_t ws_size,
                              hipStream_t stream) {
    (void)n_in; (void)out_size; (void)ws_size;
    const float* x   = (const float*)d_in[0];
    const float* gw1 = (const float*)d_in[1];
    const float* gw2 = (const float*)d_in[2];
    const float* wg  = (const float*)d_in[3];
    const float* wu  = (const float*)d_in[4];
    const float* wd  = (const float*)d_in[5];

    const int E = E_NUM;
    const int D = in_sizes[1] / E;                       // 2048
    const int N = (int)((long long)in_sizes[0] / D);     // 4096
    const int H = (int)((long long)in_sizes[3] / ((long long)E * D)); // 1024

    // workspace carve-up (256B aligned)
    size_t off = 0;
    auto carve = [&](size_t bytes) { size_t r = off; off = (off + bytes + 255) & ~(size_t)255; return r; };
    char* ws = (char*)d_ws;
    unsigned short* xb   = (unsigned short*)(ws + carve((size_t)N * D * 2));
    unsigned short* wgT  = (unsigned short*)(ws + carve((size_t)E * H * D * 2));
    unsigned short* wuT  = (unsigned short*)(ws + carve((size_t)E * H * D * 2));
    unsigned short* wdT  = (unsigned short*)(ws + carve((size_t)E * D * H * 2));
    unsigned short* hbuf = (unsigned short*)(ws + carve((size_t)2 * N * H * 2));
    float* out2          = (float*)(ws + carve((size_t)2 * N * D * 4));
    float* scoreA        = (float*)(ws + carve((size_t)N * 2 * 4));
    int*   idxA          = (int*)(ws + carve((size_t)N * 2 * 4));
    int*   ptok          = (int*)(ws + carve((size_t)2 * N * 4));
    int*   pslot         = (int*)(ws + carve((size_t)2 * N * 4));
    float* pscore        = (float*)(ws + carve((size_t)2 * N * 4));
    int*   cnt           = (int*)(ws + carve(64));
    int*   offs          = (int*)(ws + carve(64));
    int*   cursor        = (int*)(ws + carve(64));

    float* out  = (float*)d_out;
    float* lossp = out + (size_t)N * D;

    // 0) zero counters
    k_zero_i32<<<1, 32, 0, stream>>>(cnt, E);
    // 1) x -> bf16
    {
        long long n4 = (long long)N * D / 4;
        k_cvt_bf16<<<(unsigned)((n4 + 255) / 256), 256, 0, stream>>>(x, xb, n4);
    }
    // 2) weight transposes to K-contiguous bf16
    k_transpose_bf16<<<dim3(H / 32, D / 32, E), dim3(32, 8), 0, stream>>>(wg, wgT, D, H);
    k_transpose_bf16<<<dim3(H / 32, D / 32, E), dim3(32, 8), 0, stream>>>(wu, wuT, D, H);
    k_transpose_bf16<<<dim3(D / 32, H / 32, E), dim3(32, 8), 0, stream>>>(wd, wdT, H, D);
    // 3) gating
    k_gate<<<(N + 255) / 256, 256, 0, stream>>>(x, gw1, gw2, scoreA, idxA, cnt, N, D);
    // 4) offsets + compaction
    k_offsets<<<1, 32, 0, stream>>>(cnt, offs, cursor);
    k_fill<<<(N + 255) / 256, 256, 0, stream>>>(scoreA, idxA, cursor, ptok, pslot, pscore, N);
    // 5) grouped GLU GEMM (WMMA bf16 + async-to-LDS double buffering)
    k_glu_gemm<<<dim3(H / TN, (N + TM - 1) / TM, E), 256, 0, stream>>>(
        xb, wgT, wuT, ptok, cnt, offs, hbuf, D, H);
    // 6) grouped down-proj GEMM (WMMA bf16), scaled scatter to (token, slot)
    k_down_gemm<<<dim3(D / TN, (N + TM - 1) / TM, E), 256, 0, stream>>>(
        hbuf, wdT, ptok, pslot, pscore, cnt, offs, out2, D, H);
    // 7) deterministic combine of the two slots
    {
        long long total = (long long)N * (D / 4);
        k_combine<<<(unsigned)((total + 255) / 256), 256, 0, stream>>>(out2, out, N, D);
    }
    // 8) balance loss (deterministic tree reduction)
    k_loss<<<1, 256, 0, stream>>>(scoreA, idxA, cnt, lossp, N);
}
